// HGAT_77137612636491
// MI455X (gfx1250) — matrix operand order repaired and checked by reading, not compiled, not run on hardware
//
#include <hip/hip_runtime.h>

// ---------------- CDNA5 fused HGAT kernel -----------------------------------
// H=256, C=3, 3 layers, B=65536. Softmax over a singleton axis == 1.0, so the
// p-update attention degenerates to p = (xx@Wv1 + bv1) + 2p; q/k GEMMs fold
// into 256-vectors u_q = Wq@Ws[:H], u_k = Wk@Ws[H:]. Remaining work:
// per row 15 matvecs vs 256x256 weights => batched GEMM with
// v_wmma_f32_16x16x32_f16, activations register-resident per wave, weight
// slices streamed into LDS by the Tensor Data Mover (TENSORcnt-tracked).

#define Hdim 256
#define NLYR 3
#define WAVES_PB 4
#define TPB (WAVES_PB * 32)
#define ROWS_PB (WAVES_PB * 16)

typedef __attribute__((ext_vector_type(16))) _Float16 v16h;
typedef __attribute__((ext_vector_type(8)))  _Float16 v8h;
typedef __attribute__((ext_vector_type(8)))  float    v8f;
typedef __attribute__((ext_vector_type(4)))  float    f32x4;
typedef __attribute__((ext_vector_type(4)))  unsigned int u32x4;
typedef __attribute__((ext_vector_type(8)))  unsigned int u32x8;

static __device__ __forceinline__ v8f wmma_f16(v16h a, v16h b, v8f c) {
  // (neg_a, A, neg_b, B, c_mod, C, reuse_a, reuse_b)
  return __builtin_amdgcn_wmma_f32_16x16x32_f16(false, a, false, b, (short)0, c,
                                                false, false);
}

// returns b + 2*a (f16, elementwise)
static __device__ __forceinline__ v16h res2(v16h a, v16h b) {
  v16h r;
#pragma unroll
  for (int i = 0; i < 16; ++i) r[i] = b[i] + a[i] * (_Float16)2.0f;
  return r;
}

// ---------------- prep kernels ----------------------------------------------

// convert Wv[l][0] (l=0..2), Wv[l][1] (l=0..2), W1 -> f16 row-major, packed.
__global__ void prep_convert(const float* __restrict__ Wv,
                             const float* __restrict__ W1,
                             _Float16* __restrict__ Wh) {
  int idx = blockIdx.x * 256 + threadIdx.x;  // 7*65536 total
  int mat = idx >> 16;
  int off = idx & 65535;
  float v;
  if (mat < 3)      v = Wv[((size_t)(mat * 2 + 0) << 16) + off];
  else if (mat < 6) v = Wv[((size_t)((mat - 3) * 2 + 1) << 16) + off];
  else              v = W1[off];
  Wh[idx] = (_Float16)v;
}

// u_q[l][k] = sum_n Wq[l,0,k,n]*Ws[l,0,n]; u_k likewise with Ws[l,0,H+n].
// cadd[l] = bq.wsa + bk.wsb + bs[l,0]
__global__ void prep_uvec(const float* __restrict__ Wq, const float* __restrict__ Wk,
                          const float* __restrict__ bq, const float* __restrict__ bk,
                          const float* __restrict__ Ws, const float* __restrict__ bs,
                          float* __restrict__ uq, float* __restrict__ uk,
                          float* __restrict__ cadd) {
  int l = blockIdx.x, k = threadIdx.x;
  const float* wq  = Wq + ((size_t)(l * 2) << 16);
  const float* wk  = Wk + ((size_t)(l * 2) << 16);
  const float* wsa = Ws + (size_t)(l * 2) * (2 * Hdim);
  const float* wsb = wsa + Hdim;
  float s1 = 0.f, s2 = 0.f;
  for (int n = 0; n < Hdim; ++n) {
    s1 += wq[k * Hdim + n] * wsa[n];
    s2 += wk[k * Hdim + n] * wsb[n];
  }
  uq[l * Hdim + k] = s1;
  uk[l * Hdim + k] = s2;
  if (k == 0) {
    const float* bqv = bq + (size_t)(l * 2) * Hdim;
    const float* bkv = bk + (size_t)(l * 2) * Hdim;
    float c = 0.f;
    for (int n = 0; n < Hdim; ++n) c += bqv[n] * wsa[n] + bkv[n] * wsb[n];
    cadd[l] = c + bs[l * 2];
  }
}

// p0 = emb @ W_rel + b_rel  (3 x 256), stored f16
__global__ void prep_p0(const float* __restrict__ emb, const float* __restrict__ Wrel,
                        const float* __restrict__ brel, _Float16* __restrict__ p0h) {
  int c = blockIdx.x, k = threadIdx.x;
  float s = brel[k];
  for (int n = 0; n < Hdim; ++n) s += emb[c * Hdim + n] * Wrel[n * Hdim + k];
  p0h[c * Hdim + k] = (_Float16)s;
}

// ---------------- fused main kernel -----------------------------------------

__global__ __launch_bounds__(TPB, 1) void hgat_fused(
    const float* __restrict__ x, const _Float16* __restrict__ Wh,
    const _Float16* __restrict__ p0h, const float* __restrict__ uq,
    const float* __restrict__ uk, const float* __restrict__ cadd,
    const float* __restrict__ bv, const float* __restrict__ b1,
    const float* __restrict__ W2, const float* __restrict__ b2,
    float* __restrict__ out) {
  // LDS: double-buffered 256x16 f16 weight slice (TDM-filled) + per-wave
  // 16x256 transpose scratch (column-major: scr[n*16 + row]).
  __shared__ __attribute__((aligned(32))) _Float16 s_slice[2][256 * 16];  // 16KB
  __shared__ __attribute__((aligned(16))) _Float16 s_scr[WAVES_PB][16 * 256];  // 32KB

  const int tid  = threadIdx.x;
  const int wave = tid >> 5;
  const int lane = tid & 31;
  const int ml   = lane & 15;   // row (A/D col) index within tile
  const int hf   = lane >> 4;   // half-wave
  const long rowbase = (long)blockIdx.x * ROWS_PB + (long)wave * 16;
  const long myrow   = rowbase + ml;

  v16h xxA[8];      // xx as A fragments, K-step j covers K=32j..32j+31
  v16h pA[3][8];    // p channels as A fragments

  // -- init xx from x (f32 -> f16), ISA A-layout: chunks at 32j+8hf, +16
#pragma unroll
  for (int j = 0; j < 8; ++j) {
    const int c0 = 32 * j + 8 * hf;
    const float* px = x + (size_t)myrow * Hdim;
    f32x4 a0 = *(const f32x4*)(px + c0);
    f32x4 a1 = *(const f32x4*)(px + c0 + 4);
    f32x4 b0 = *(const f32x4*)(px + c0 + 16);
    f32x4 b1l = *(const f32x4*)(px + c0 + 20);
    v16h f;
#pragma unroll
    for (int i = 0; i < 4; ++i) {
      f[i]      = (_Float16)a0[i];
      f[4 + i]  = (_Float16)a1[i];
      f[8 + i]  = (_Float16)b0[i];
      f[12 + i] = (_Float16)b1l[i];
    }
    xxA[j] = f;
  }
  // -- init p from p0h (same for every row; layout still valid)
#pragma unroll
  for (int c = 0; c < 3; ++c)
#pragma unroll
    for (int j = 0; j < 8; ++j) {
      const int c0 = 32 * j + 8 * hf;
      v8h lo = *(const v8h*)(p0h + c * Hdim + c0);
      v8h hi = *(const v8h*)(p0h + c * Hdim + c0 + 16);
      pA[c][j] = __builtin_shufflevector(lo, hi, 0, 1, 2, 3, 4, 5, 6, 7, 8, 9,
                                         10, 11, 12, 13, 14, 15);
    }

  // per-row dot of an A-fragment set with a shared f32 vector
  auto dot_u = [&](const v16h* fr, const float* u) -> float {
    float s = 0.f;
#pragma unroll
    for (int j = 0; j < 8; ++j) {
      const int c0 = 32 * j + 8 * hf;
      f32x4 u0 = *(const f32x4*)(u + c0);
      f32x4 u1 = *(const f32x4*)(u + c0 + 4);
      f32x4 u2 = *(const f32x4*)(u + c0 + 16);
      f32x4 u3 = *(const f32x4*)(u + c0 + 20);
#pragma unroll
      for (int i = 0; i < 4; ++i) {
        s += (float)fr[j][i] * u0[i];
        s += (float)fr[j][4 + i] * u1[i];
        s += (float)fr[j][8 + i] * u2[i];
        s += (float)fr[j][12 + i] * u3[i];
      }
    }
    s += __shfl_xor(s, 16, 32);  // combine the two half-wave K partials
    return s;
  };

  // ---- Tensor Data Mover stage: one TDM op pulls the 256x16 f16 slice
  // (256 rows x 32B from a row-major 256x512B tensor) into LDS, packed.
  // D# group0: count=1, lds_addr, 57-bit global addr, type=2.
  // D# group1: data_size=4B, tensor_dim0=128u, tile_dim0=8u, tile_dim1=256,
  //            tensor_dim0_stride=128u, no padding / iterate / multicast.
  auto stage = [&](const _Float16* Wg, int nt, int buf) {
    if (wave == 0) {
      unsigned long long ga =
          (unsigned long long)(size_t)Wg + (unsigned long long)nt * 32ull;
      unsigned lds = (unsigned)(size_t)(&s_slice[buf][0]);
      u32x4 g0;
      g0[0] = 1u;                                  // count=1 (valid user D#)
      g0[1] = lds;                                 // lds_addr
      g0[2] = (unsigned)ga;                        // global_addr[31:0]
      g0[3] = (unsigned)(ga >> 32) | (2u << 30);   // global_addr[56:32]|type=2
      u32x8 g1;
      g1[0] = (2u << 16);                 // workgroup_mask=0, data_size=4B
      g1[1] = (128u << 16);               // tensor_dim0[15:0] = 128 units
      g1[2] = (256u << 16);               // tensor_dim0 hi=0 | tensor_dim1 lo
      g1[3] = (8u << 16);                 // tensor_dim1 hi=0 | tile_dim0=8
      g1[4] = 256u;                       // tile_dim1=256, tile_dim2=0
      g1[5] = 128u;                       // tensor_dim0_stride lo = 128 units
      g1[6] = 0u;                         // stride0 hi | tensor_dim1_stride lo
      g1[7] = 0u;                         // tensor_dim1_stride hi (unused, 2D)
      asm volatile("tensor_load_to_lds %0, %1" ::"s"(g0), "s"(g1) : "memory");
    }
  };
  // wait for this wave's outstanding TDM ops, then publish to the block
  auto stage_wait = [&]() {
    if (wave == 0) __builtin_amdgcn_s_wait_tensorcnt(0);
    __syncthreads();
  };

  // B fragment: lane holds column n = ml, K chunks at 32j+8hf, +16 (A-matching)
  auto bfrag = [&](int buf, int j) -> v16h {
    v16h r;
    const int c0 = 32 * j + 8 * hf;
    const _Float16* sl = &s_slice[buf][0];
#pragma unroll
    for (int i = 0; i < 8; ++i) {
      r[i]     = sl[(c0 + i) * 16 + ml];
      r[8 + i] = sl[(c0 + 16 + i) * 16 + ml];
    }
    return r;
  };

  // D tile -> per-wave scratch, column-major packed (one b128 store per lane)
  auto scr_store = [&](int nt, v8h pk) {
    *(v8h*)(&s_scr[wave][(nt * 16 + ml) * 16 + 8 * hf]) = pk;
  };
  // scratch -> A fragment (transpose gather)
  auto scr_loadA = [&](int j) -> v16h {
    v16h r;
    const int c0 = 32 * j + 8 * hf;
    const _Float16* sc = &s_scr[wave][0];
#pragma unroll
    for (int i = 0; i < 8; ++i) {
      r[i]     = sc[(c0 + i) * 16 + ml];
      r[8 + i] = sc[(c0 + 16 + i) * 16 + ml];
    }
    return r;
  };

  const float b2v = b2[0];
  float opart[3][8];
#pragma unroll
  for (int c = 0; c < 3; ++c)
#pragma unroll
    for (int r = 0; r < 8; ++r) opart[c][r] = 0.f;

  for (int l = 0; l < NLYR; ++l) {
    // ---- attention scores (q/k GEMMs folded into u-vectors)
    float sq  = dot_u(xxA, uq + l * Hdim);
    float sk0 = dot_u(pA[0], uk + l * Hdim);
    float sk1 = dot_u(pA[1], uk + l * Hdim);
    float sk2 = dot_u(pA[2], uk + l * Hdim);
    float cl = cadd[l];
    float s0 = sq + sk0 + cl, s1 = sq + sk1 + cl, s2 = sq + sk2 + cl;
    s0 = s0 > 0.f ? s0 : 0.2f * s0;
    s1 = s1 > 0.f ? s1 : 0.2f * s1;
    s2 = s2 > 0.f ? s2 : 0.2f * s2;
    float mx = fmaxf(s0, fmaxf(s1, s2));
    float e0 = __expf(s0 - mx), e1 = __expf(s1 - mx), e2 = __expf(s2 - mx);
    float inv = 1.f / (e0 + e1 + e2);
    float w0 = e0 * inv, w1 = e1 * inv, w2 = e2 * inv;
    // distribute per-row weights to the D-fragment row each (lane,reg) owns
    float wr0[8], wr1[8], wr2[8];
#pragma unroll
    for (int r = 0; r < 8; ++r) {
      int src = r + 8 * hf;
      wr0[r] = __shfl(w0, src, 32);
      wr1[r] = __shfl(w1, src, 32);
      wr2[r] = __shfl(w2, src, 32);
    }

    // ---- GEMM1: v_c = p_c @ Wv[l][0] + bv[l][0]; xx = sum_c w_c v_c + 2 xx
    {
      const _Float16* Wg = Wh + ((size_t)l << 16);
      const float* bias = bv + (size_t)(l * 2 + 0) * Hdim;
      stage(Wg, 0, 0);
      float bcur = bias[ml];  // bias for tile 0, pipelined one tile ahead
      stage_wait();
      for (int nt = 0; nt < 16; ++nt) {
        int cur = nt & 1;
        if (nt < 15) stage(Wg, nt + 1, cur ^ 1);
        int ntn = nt < 15 ? nt + 1 : 15;
        float bnxt = bias[ntn * 16 + ml];  // hides under this tile's WMMAs
        v8f a0, a1, a2;
#pragma unroll
        for (int r = 0; r < 8; ++r) { a0[r] = bcur; a1[r] = bcur; a2[r] = bcur; }
        // 2-deep B-fragment pipeline: keep 2 fragments' strided LDS gathers in flight
        v16h bf0 = bfrag(cur, 0);
        v16h bf1 = bfrag(cur, 1);
#pragma unroll
        for (int j = 0; j < 8; ++j) {
          v16h bfn = bfrag(cur, (j + 2 < 8) ? j + 2 : 7);
          a0 = wmma_f16(pA[0][j], bf0, a0);
          a1 = wmma_f16(pA[1][j], bf0, a1);
          a2 = wmma_f16(pA[2][j], bf0, a2);
          bf0 = bf1;
          bf1 = bfn;
        }
        v8h pk;
#pragma unroll
        for (int r = 0; r < 8; ++r)
          pk[r] = (_Float16)(wr0[r] * a0[r] + wr1[r] * a1[r] + wr2[r] * a2[r]);
        scr_store(nt, pk);
        bcur = bnxt;
        stage_wait();
      }
#pragma unroll
      for (int j = 0; j < 8; ++j) {
        v16h at = scr_loadA(j);
        xxA[j] = res2(xxA[j], at);  // xx = attn + 2*xx
      }
      __syncthreads();
    }

    // ---- GEMM2: v2 = xx @ Wv[l][1] + bv[l][1]; p_c = v2 + 2 p_c
    {
      const _Float16* Wg = Wh + ((size_t)(3 + l) << 16);
      const float* bias = bv + (size_t)(l * 2 + 1) * Hdim;
      stage(Wg, 0, 0);
      float bcur = bias[ml];
      stage_wait();
      for (int nt = 0; nt < 16; ++nt) {
        int cur = nt & 1;
        if (nt < 15) stage(Wg, nt + 1, cur ^ 1);
        int ntn = nt < 15 ? nt + 1 : 15;
        float bnxt = bias[ntn * 16 + ml];
        v8f a0;
#pragma unroll
        for (int r = 0; r < 8; ++r) a0[r] = bcur;
        v16h bf0 = bfrag(cur, 0);
        v16h bf1 = bfrag(cur, 1);
#pragma unroll
        for (int j = 0; j < 8; ++j) {
          v16h bfn = bfrag(cur, (j + 2 < 8) ? j + 2 : 7);
          a0 = wmma_f16(xxA[j], bf0, a0);
          bf0 = bf1;
          bf1 = bfn;
        }
        v8h pk;
#pragma unroll
        for (int r = 0; r < 8; ++r) pk[r] = (_Float16)a0[r];
        scr_store(nt, pk);
        bcur = bnxt;
        stage_wait();
      }
#pragma unroll
      for (int j = 0; j < 8; ++j) {
        v16h v2 = scr_loadA(j);
#pragma unroll
        for (int c = 0; c < 3; ++c) pA[c][j] = res2(pA[c][j], v2);
      }
      __syncthreads();
    }
  }

  // ---- final: h = tanh(p @ W1 + b1); out = sigmoid(h . W2 + b2)
  {
    const _Float16* Wg = Wh + ((size_t)6 << 16);
    stage(Wg, 0, 0);
    float bcur = b1[ml];
    float wcur = W2[ml];
    stage_wait();
    for (int nt = 0; nt < 16; ++nt) {
      int cur = nt & 1;
      if (nt < 15) stage(Wg, nt + 1, cur ^ 1);
      int ntn = nt < 15 ? nt + 1 : 15;
      float bnxt = b1[ntn * 16 + ml];
      float wnxt = W2[ntn * 16 + ml];
      v8f a0, a1, a2;
#pragma unroll
      for (int r = 0; r < 8; ++r) { a0[r] = bcur; a1[r] = bcur; a2[r] = bcur; }
      v16h bf0 = bfrag(cur, 0);
      v16h bf1 = bfrag(cur, 1);
#pragma unroll
      for (int j = 0; j < 8; ++j) {
        v16h bfn = bfrag(cur, (j + 2 < 8) ? j + 2 : 7);
        a0 = wmma_f16(pA[0][j], bf0, a0);
        a1 = wmma_f16(pA[1][j], bf0, a1);
        a2 = wmma_f16(pA[2][j], bf0, a2);
        bf0 = bf1;
        bf1 = bfn;
      }
#pragma unroll
      for (int r = 0; r < 8; ++r) {
        opart[0][r] += tanhf(a0[r]) * wcur;
        opart[1][r] += tanhf(a1[r]) * wcur;
        opart[2][r] += tanhf(a2[r]) * wcur;
      }
      bcur = bnxt;
      wcur = wnxt;
      stage_wait();
    }
  }
  // reduce the hidden-dim partials across the 16 lanes of each half-wave
#pragma unroll
  for (int c = 0; c < 3; ++c)
#pragma unroll
    for (int r = 0; r < 8; ++r) {
      float v = opart[c][r];
      v += __shfl_xor(v, 1, 32);
      v += __shfl_xor(v, 2, 32);
      v += __shfl_xor(v, 4, 32);
      v += __shfl_xor(v, 8, 32);
      if (ml == 0) {
        long row = rowbase + 8 * hf + r;
        out[row * 3 + c] = 1.f / (1.f + __expf(-(v + b2v)));
      }
    }
}

// ---------------- launch ----------------------------------------------------

extern "C" void kernel_launch(void* const* d_in, const int* in_sizes, int n_in,
                              void* d_out, int out_size, void* d_ws,
                              size_t ws_size, hipStream_t stream) {
  const float* x    = (const float*)d_in[0];
  const float* emb  = (const float*)d_in[1];
  const float* Wrel = (const float*)d_in[2];
  const float* brel = (const float*)d_in[3];
  const float* Wq   = (const float*)d_in[4];
  const float* bq   = (const float*)d_in[5];
  const float* Wk   = (const float*)d_in[6];
  const float* bk   = (const float*)d_in[7];
  const float* Wv   = (const float*)d_in[8];
  const float* bv   = (const float*)d_in[9];
  const float* Ws   = (const float*)d_in[10];
  const float* bs   = (const float*)d_in[11];
  const float* W1   = (const float*)d_in[12];
  const float* b1   = (const float*)d_in[13];
  const float* W2   = (const float*)d_in[14];
  const float* b2   = (const float*)d_in[15];
  // d_in[16] = mask: provably irrelevant (softmax over singleton axis == 1)
  float* out = (float*)d_out;

  _Float16* Wh  = (_Float16*)d_ws;           // 7 * 65536 f16
  _Float16* p0h = Wh + (size_t)7 * 65536;    // 3 * 256 f16
  float* uq   = (float*)(p0h + 3 * Hdim);    // 3 * 256 f32
  float* uk   = uq + 3 * Hdim;               // 3 * 256 f32
  float* cadd = uk + 3 * Hdim;               // 3 f32

  prep_convert<<<7 * 65536 / 256, 256, 0, stream>>>(Wv, W1, Wh);
  prep_uvec<<<3, 256, 0, stream>>>(Wq, Wk, bq, bk, Ws, bs, uq, uk, cadd);
  prep_p0<<<3, 256, 0, stream>>>(emb, Wrel, brel, p0h);
  hgat_fused<<<65536 / ROWS_PB, TPB, 0, stream>>>(x, Wh, p0h, uq, uk, cadd, bv,
                                                  b1, W2, b2, out);
}
